// DynamicTrackHead_47382079209634
// MI455X (gfx1250) — matrix operand order repaired and checked by reading, not compiled, not run on hardware
//
#include <hip/hip_runtime.h>
#include <hip/hip_bf16.h>
#include <stdint.h>

typedef __attribute__((ext_vector_type(16))) _Float16     v16h;
typedef __attribute__((ext_vector_type(2)))  _Float16     v2h;
typedef __attribute__((ext_vector_type(8)))  float        v8f;
typedef __attribute__((ext_vector_type(4)))  unsigned int v4u;
typedef __attribute__((ext_vector_type(8)))  int          v8i;
typedef __attribute__((ext_vector_type(4)))  int          v4i;

#define H_  96
#define W_  160
#define HW_ (H_ * W_)
#define P_  736

__device__ inline v8f wmma_f32_f16(v16h a, v16h b, v8f c) {
  return __builtin_amdgcn_wmma_f32_16x16x32_f16(false, a, false, b, (short)0, c, false, false);
}

__device__ inline int v2h_to_i(v2h x) { int r; __builtin_memcpy(&r, &x, 4); return r; }
__device__ inline v2h i_to_v2h(int x) { v2h r; __builtin_memcpy(&r, &x, 4); return r; }

// ReLU(+already-h-pooled) values -> packed f16, vertical pool across halves,
// each half stores its two channel-pair words. One cndmask per store.
__device__ inline void pool_store(const float vm[8], int half,
                                  unsigned* __restrict__ dst0, int plane_stride) {
  int w4[4];
#pragma unroll
  for (int j = 0; j < 4; ++j) {
    v2h p; p[0] = (_Float16)vm[2 * j]; p[1] = (_Float16)vm[2 * j + 1];
    const v2h q = i_to_v2h(__shfl_xor(v2h_to_i(p), 16, 32));   // vertical pool
    p = __builtin_elementwise_max(p, q);                       // v_pk_max_num_f16
    w4[j] = v2h_to_i(p);
  }
  const unsigned wlo = (unsigned)(half ? w4[2] : w4[0]);
  const unsigned whi = (unsigned)(half ? w4[3] : w4[1]);
  unsigned* d = dst0 + (half ? 2 * plane_stride : 0);
  d[0]            = wlo;
  d[plane_stride] = whi;
}

// Dual-row A fragment: rows 0..7 = W[m][k] at K 0..Krows-1,
// rows 8..15 = W[m-8][k] at K 16..16+Krows-1.
__device__ inline v16h build_a_dual(const float* W, int Krows, int lane) {
  const int m = lane & 15, half = lane >> 4;
  v16h a;
#pragma unroll
  for (int j = 0; j < 8; ++j) {
    const int k0 = 2 * j + 16 * half;
    const int row = (m < 8) ? m : (m - 8);
    const int kb  = (m < 8) ? 0 : 16;
    const int kk0 = k0 - kb, kk1 = k0 + 1 - kb;
    float lo = 0.f, hi = 0.f;
    if (kk0 >= 0 && kk0 < Krows) lo = W[row * Krows + kk0];
    if (kk1 >= 0 && kk1 < Krows) hi = W[row * Krows + kk1];
    a[2 * j]     = (_Float16)lo;
    a[2 * j + 1] = (_Float16)hi;
  }
  return a;
}

// Single A fragment (layer 3): rows mbase..mbase+15, K 0..Krows-1.
__device__ inline v16h build_a_frag(const float* W, int Krows, int Mrows, int mbase, int lane) {
  const int m = lane & 15, half = lane >> 4;
  v16h a;
#pragma unroll
  for (int j = 0; j < 8; ++j) {
    const int k0 = 2 * j + 16 * half;
    float lo = 0.f, hi = 0.f;
    const int M = mbase + m;
    if (M < Mrows) {
      if (k0 < Krows)     lo = W[M * Krows + k0];
      if (k0 + 1 < Krows) hi = W[M * Krows + k0 + 1];
    }
    a[2 * j]     = (_Float16)lo;
    a[2 * j + 1] = (_Float16)hi;
  }
  return a;
}

// TDM: DMA the 736-float param vector into LDS (1-D tile).
__device__ inline void tdm_load_params(const float* gsrc, float* ldst) {
  const unsigned lds_addr = (unsigned)(uintptr_t)ldst;
  const unsigned long long ga = (unsigned long long)(uintptr_t)gsrc;
  v4u g0;
  g0[0] = 1u;                                                 // count=1
  g0[1] = lds_addr;
  g0[2] = (unsigned)(ga & 0xFFFFFFFFu);
  g0[3] = (unsigned)((ga >> 32) & 0x01FFFFFFu) | 0x80000000u; // type=2
  v8i g1;
  g1[0] = 0x00020000;                // data_size=2 (4B)
  g1[1] = (int)(736u << 16);         // tensor_dim0
  g1[2] = (int)(1u << 16);           // tensor_dim1=1
  g1[3] = (int)(736u << 16);         // tile_dim0=736
  g1[4] = 1;                         // tile_dim1=1
  g1[5] = 736;                       // tensor_dim0_stride
  g1[6] = (int)(736u << 16);         // tensor_dim1_stride
  g1[7] = 0;
  v4i gz = {0, 0, 0, 0};
#if defined(__clang_major__) && (__clang_major__ >= 23)
  v8i gz8 = {0, 0, 0, 0, 0, 0, 0, 0};
  __builtin_amdgcn_tensor_load_to_lds(g0, g1, gz, gz, gz8, 0);
#else
  __builtin_amdgcn_tensor_load_to_lds(g0, g1, gz, gz, 0);
#endif
}

// One block per (head, instance). 8 waves of 32.
__global__ void __launch_bounds__(256)
track_head_kernel(const float* __restrict__ mask_feats,
                  const float* __restrict__ main_params,
                  const float* __restrict__ side_params,
                  const int*   __restrict__ im_inds,
                  const float* __restrict__ inst_loc,
                  const float* __restrict__ offset_pred,
                  const int*   __restrict__ fpn_levels,
                  const int*   __restrict__ stride_ptr,
                  float*       __restrict__ stage,
                  int n_inst)
{
  __shared__ float        sP[P_];                 // dynamic params (TDM target)
  __shared__ unsigned int pooled1p[4][8][96];     // [ch-pair][strip row][x] f16x2 (pad 80->96)
  __shared__ unsigned int pooled2p[4][961];       // [ch-pair][24*40 + dummy]
  __shared__ float        acc3[64];

  const int tid  = threadIdx.x;
  const int lane = tid & 31;
  const int wv   = tid >> 5;
  const int n16  = lane & 15;
  const int half = lane >> 4;

  const int bx   = blockIdx.x;
  const int head = bx / n_inst;
  const int inst = bx - head * n_inst;

  const float* params = (head == 0 ? main_params : side_params) + (size_t)inst * P_;
  if (wv == 0) {
    tdm_load_params(params, sP);
    __builtin_amdgcn_s_wait_tensorcnt(0);
  }
  if (tid < 64) acc3[tid] = 0.f;

  const int    im    = im_inds[inst];
  const float* feats = mask_feats + (size_t)im * 8 * HW_;
  float lx = inst_loc[inst * 2 + 0];
  float ly = inst_loc[inst * 2 + 1];
  if (head == 1) {
    lx += offset_pred[inst * 2 + 0] * 128.0f;
    ly += offset_pred[inst * 2 + 1] * 128.0f;
  }
  const int   fl = fpn_levels[inst];
  const float soi_tab[5] = {64.f, 128.f, 256.f, 512.f, 1024.f};
  const float inv_soi = 1.0f / soi_tab[fl];
  const int   s  = stride_ptr[0];
  const float fs = (float)s;
  const float hs = (float)(s >> 1);

  __syncthreads();

  v16h a1 = build_a_dual(sP + 0,  10, lane);     // W1 [8][10]
  v16h a2 = build_a_dual(sP + 80,  8, lane);     // W2 [8][8]
  v8f cb1, cb2;                                  // bias folded into WMMA C operand
#pragma unroll
  for (int r = 0; r < 8; ++r) { cb1[r] = sP[656 + r]; cb2[r] = sP[664 + r]; }

  // Strips of 16 input rows: 8 row-pairs -> 8 pooled1 rows -> 4 pooled2 rows
  for (int t = 0; t < 6; ++t) {
    // ---- Layer 1 + 2x2 pool. Task = (row-pair, 32-px tile): 2 WMMAs (even/odd x).
    for (int task = wv; task < 40; task += 8) {
      const int rp  = task / 5;                  // local row-pair 0..7
      const int tx  = task - rp * 5;
      const int x0  = tx * 32;
      const int row = 2 * (8 * t + rp) + half;   // each half handles its own row
      const int xe  = x0 + 2 * n16;
      const float gy   = (float)row * fs + hs;
      const float rely = (ly - gy) * inv_soi;
      const float* fpe = feats + row * W_ + xe;
      __builtin_prefetch(fpe + 2 * W_, 0, 0);

      v16h be, bo;
#pragma unroll
      for (int i = 0; i < 16; ++i) { be[i] = (_Float16)0.f; bo[i] = (_Float16)0.f; }
      const float gxe = (float)xe * fs + hs;
      be[0] = (_Float16)((lx - gxe) * inv_soi);
      be[1] = (_Float16)rely;
      bo[0] = (_Float16)((lx - gxe - fs) * inv_soi);
      bo[1] = (_Float16)rely;
#pragma unroll
      for (int c = 0; c < 8; ++c) {              // 8-byte aligned pair loads
        const float2 f2 = *reinterpret_cast<const float2*>(fpe + c * HW_);
        be[2 + c] = (_Float16)f2.x;
        bo[2 + c] = (_Float16)f2.y;
      }

      v8f de = wmma_f32_f16(a1, be, cb1);        // lanes 0-15 row0, 16-31 row1
      v8f dd = wmma_f32_f16(a1, bo, cb1);
      float vm[8];
#pragma unroll
      for (int r = 0; r < 8; ++r)
        vm[r] = fmaxf(fmaxf(de[r], dd[r]), 0.f); // ReLU + horizontal pool in-lane

      const int xout = (x0 >> 1) + n16;          // 0..79
      pool_store(vm, half, &pooled1p[0][rp][xout], 8 * 96);
    }
    __syncthreads();

    // ---- Layer 2 + 2x2 pool: pooled1 rows 0..7 -> pooled2 rows 4t..4t+3.
    for (int task = wv; task < 12; task += 8) {
      const int rp   = task / 3;                 // layer2 row-pair 0..3
      const int tx   = task - rp * 3;
      const int x0o  = tx * 16;                  // output x tile base (0,16,32)
      const int rowl = 2 * rp + half;
      const int xe   = 2 * (x0o + n16);          // input x (may touch pad region)

      v16h be, bo;
#pragma unroll
      for (int i = 0; i < 16; ++i) { be[i] = (_Float16)0.f; bo[i] = (_Float16)0.f; }
#pragma unroll
      for (int j = 0; j < 4; ++j) {              // 8-byte aligned even/odd pair
        const uint2 pp = *reinterpret_cast<const uint2*>(&pooled1p[j][rowl][xe]);
        const v2h pe = i_to_v2h((int)pp.x);
        const v2h po = i_to_v2h((int)pp.y);
        be[2 * j] = pe[0]; be[2 * j + 1] = pe[1];
        bo[2 * j] = po[0]; bo[2 * j + 1] = po[1];
      }

      v8f de = wmma_f32_f16(a2, be, cb2);
      v8f dd = wmma_f32_f16(a2, bo, cb2);
      float vm[8];
#pragma unroll
      for (int r = 0; r < 8; ++r)
        vm[r] = fmaxf(fmaxf(de[r], dd[r]), 0.f);

      const int xout = x0o + n16;                // 0..47; valid < 40
      const int idx  = (xout < 40) ? ((4 * t + rp) * 40 + xout) : 960;  // dummy slot
      pool_store(vm, half, &pooled2p[0][idx], 961);
    }
    __syncthreads();
  }

  // ---- Layer 3: W3[64][8] @ pooled2[8][960], per-pixel ReLU, then mean.
  {
    const int rt = wv & 3;
    v16h a3 = build_a_frag(sP + 144, 8, 64, rt * 16, lane);
    v8f cb3;
#pragma unroll
    for (int r = 0; r < 8; ++r) cb3[r] = sP[672 + rt * 16 + half * 8 + r];
    float acc[8];
#pragma unroll
    for (int r = 0; r < 8; ++r) acc[r] = 0.f;

    for (int pt = (wv >> 2); pt < 60; pt += 2) {
      const int q = pt * 16 + n16;
      v16h b;
#pragma unroll
      for (int i = 0; i < 16; ++i) b[i] = (_Float16)0.f;
#pragma unroll
      for (int j = 0; j < 4; ++j) {              // A rows K>=16 are zero: half1 junk is inert
        const v2h p = i_to_v2h((int)pooled2p[j][q]);
        b[2 * j] = p[0]; b[2 * j + 1] = p[1];
      }
      v8f d = wmma_f32_f16(a3, b, cb3);
#pragma unroll
      for (int r = 0; r < 8; ++r)
        acc[r] += fmaxf(0.f, d[r]);              // ReLU before mean (bias in C)
    }
#pragma unroll
    for (int r = 0; r < 8; ++r) {
#pragma unroll
      for (int off = 1; off < 16; off <<= 1)
        acc[r] += __shfl_xor(acc[r], off, 32);
      if (n16 == 0)
        atomicAdd(&acc3[rt * 16 + half * 8 + r], acc[r]);
    }
  }
  __syncthreads();
  if (tid < 64)
    stage[(size_t)bx * 64 + tid] = acc3[tid] * (1.0f / 960.0f);
}

// Per (head, instance) MLP: 64 -> 128 (LayerNorm, SELU) -> 64
__global__ void __launch_bounds__(128)
mlp_kernel(const float* __restrict__ stage,
           const float* __restrict__ mw1, const float* __restrict__ mb1,
           const float* __restrict__ mg,  const float* __restrict__ mbeta,
           const float* __restrict__ mw2, const float* __restrict__ mb2,
           const float* __restrict__ sw1, const float* __restrict__ sb1,
           const float* __restrict__ sg,  const float* __restrict__ sbeta,
           const float* __restrict__ sw2, const float* __restrict__ sb2,
           float* __restrict__ out, int n_inst)
{
  __shared__ float v[64];
  __shared__ float hbuf[128];
  __shared__ float tmp[128];
  __shared__ float red[2];
  const int tid  = threadIdx.x;
  const int b    = blockIdx.x;
  const int head = b / n_inst;

  const float* w1 = head ? sw1   : mw1;
  const float* b1 = head ? sb1   : mb1;
  const float* g  = head ? sg    : mg;
  const float* be = head ? sbeta : mbeta;
  const float* w2 = head ? sw2   : mw2;
  const float* b2 = head ? sb2   : mb2;

  if (tid < 64) v[tid] = stage[(size_t)b * 64 + tid];
  __syncthreads();

  float hj = b1[tid];
  const float* wrow = w1 + tid * 64;
#pragma unroll 8
  for (int k = 0; k < 64; ++k) hj += v[k] * wrow[k];

  tmp[tid] = hj; __syncthreads();
  for (int off = 64; off > 0; off >>= 1) {
    if (tid < off) tmp[tid] += tmp[tid + off];
    __syncthreads();
  }
  if (tid == 0) red[0] = tmp[0] * (1.0f / 128.0f);
  __syncthreads();
  const float mu = red[0];
  const float dv = hj - mu;
  tmp[tid] = dv * dv; __syncthreads();
  for (int off = 64; off > 0; off >>= 1) {
    if (tid < off) tmp[tid] += tmp[tid + off];
    __syncthreads();
  }
  if (tid == 0) red[1] = tmp[0] * (1.0f / 128.0f);
  __syncthreads();
  const float var = red[1];
  const float xn  = dv * rsqrtf(var + 1e-5f) * g[tid] + be[tid];
  const float alpha = 1.6732632423543772f, scl = 1.0507009873554805f;
  hbuf[tid] = (xn > 0.f) ? (scl * xn) : (scl * alpha * (expf(xn) - 1.0f));
  __syncthreads();

  if (tid < 64) {
    float o = b2[tid];
    const float* w2r = w2 + tid * 128;
#pragma unroll 8
    for (int j = 0; j < 128; ++j) o += hbuf[j] * w2r[j];
    out[(size_t)b * 64 + tid] = o;
  }
}

extern "C" void kernel_launch(void* const* d_in, const int* in_sizes, int n_in,
                              void* d_out, int out_size, void* d_ws, size_t ws_size,
                              hipStream_t stream) {
  const float* mask_feats = (const float*)d_in[0];
  const float* mparams    = (const float*)d_in[1];
  const float* sparams    = (const float*)d_in[2];
  const int*   im_inds    = (const int*)  d_in[3];
  const float* iloc       = (const float*)d_in[4];
  const float* opred      = (const float*)d_in[5];
  const int*   fpn        = (const int*)  d_in[6];
  const float* mw1        = (const float*)d_in[7];
  const float* mb1        = (const float*)d_in[8];
  const float* mg         = (const float*)d_in[9];
  const float* mbeta      = (const float*)d_in[10];
  const float* mw2        = (const float*)d_in[11];
  const float* mb2        = (const float*)d_in[12];
  const float* sw1        = (const float*)d_in[13];
  const float* sb1        = (const float*)d_in[14];
  const float* sg         = (const float*)d_in[15];
  const float* sbeta      = (const float*)d_in[16];
  const float* sw2        = (const float*)d_in[17];
  const float* sb2        = (const float*)d_in[18];
  const int*   stride_ptr = (const int*)  d_in[19];

  const int n_inst = in_sizes[3];
  float* stage = (float*)d_ws;        // [2*n_inst, 64]
  float* out   = (float*)d_out;       // main_logits ++ side_logits

  track_head_kernel<<<2 * n_inst, 256, 0, stream>>>(
      mask_feats, mparams, sparams, im_inds, iloc, opred, fpn, stride_ptr,
      stage, n_inst);

  mlp_kernel<<<2 * n_inst, 128, 0, stream>>>(
      stage, mw1, mb1, mg, mbeta, mw2, mb2,
      sw1, sb1, sg, sbeta, sw2, sb2, out, n_inst);
}
